// VQEmbedding_11519102288009
// MI455X (gfx1250) — matrix operand order repaired and checked by reading, not compiled, not run on hardware
//
#include <hip/hip_runtime.h>

typedef __attribute__((ext_vector_type(16))) __bf16 v16bf;
typedef __attribute__((ext_vector_type(8)))  __bf16 bf16x8;
typedef __attribute__((ext_vector_type(8)))  float  v8f;

#define NUM_EMB   1024
#define EMB_DIM   64
#define HW        4096          // 64*64 spatial
#define NPOS      131072        // 32 * 4096 rows
#define NUMEL     8388608       // 32*64*4096
#define LOSS_W    1.25f         // 1.0 recon + 0.25 commitment (same forward value)
#define NT        (NUM_EMB / 16)  // 64 codebook tiles

#define ROW_STRIDE 72                                   // padded bf16 row in LDS (144B)
#define LDS_EMB_BYTES  (NUM_EMB * ROW_STRIDE * 2)       // 147456
#define LDS_NORM_BYTES (NUM_EMB * 4)                    // 4096
#define WAVES_PER_BLOCK 8
#define MT_PER_WAVE 4
#define ROWS_PER_BLOCK (WAVES_PER_BLOCK * MT_PER_WAVE * 16)  // 512
#define GRID1 (NPOS / ROWS_PER_BLOCK)                        // 256 blocks

union ABu { bf16x8 h[2]; v16bf v; };

__global__ __launch_bounds__(256, 1)
void vq_fused_kernel(const float* __restrict__ z_e,
                     const float* __restrict__ emb,
                     float* __restrict__ z_q_out,
                     int* __restrict__ idx_out,
                     float* __restrict__ partials)
{
  extern __shared__ char smem[];
  __bf16* lds_emb  = (__bf16*)smem;
  float*  lds_norm = (float*)(smem + LDS_EMB_BYTES);
  int*    lds_idx  = (int*)(smem + LDS_EMB_BYTES + LDS_NORM_BYTES);
  float*  lds_red  = (float*)(smem + LDS_EMB_BYTES + LDS_NORM_BYTES +
                              WAVES_PER_BLOCK * 16 * 4);

  const int tid = threadIdx.x;

  // ---- stage codebook into LDS as bf16 (padded rows) + fp32 row norms ----
  for (int i = tid; i < NUM_EMB * EMB_DIM; i += 256) {
    const int n = i >> 6, c = i & 63;
    lds_emb[n * ROW_STRIDE + c] = (__bf16)emb[i];
  }
  for (int n = tid; n < NUM_EMB; n += 256) {
    const float* r = emb + n * EMB_DIM;
    float s = 0.f;
    #pragma unroll
    for (int c = 0; c < EMB_DIM; ++c) s += r[c] * r[c];
    lds_norm[n] = s;
  }
  __syncthreads();

  const int lane = tid & 31;
  const int wave = tid >> 5;
  const int hf   = lane >> 4;   // which 16-lane half
  const int l16  = lane & 15;

  float lsum = 0.f;

  for (int j = 0; j < MT_PER_WAVE; ++j) {
    const int mtile   = (blockIdx.x * WAVES_PER_BLOCK + wave) * MT_PER_WAVE + j;
    const int rowbase = mtile * 16;
    const int bidx    = rowbase >> 12;            // batch
    const int p       = (rowbase & 4095) + l16;   // spatial position
    const size_t zoff = (size_t)bidx * (EMB_DIM * HW) + p;
    const float* zb   = z_e + zoff;

    // ---- load A tile (16 rows x 64 ch): keep fp32 for exact loss; build
    //      bf16 fragments per the 16-bit A 16x32 layout:
    //      lane<16: K in {0..7, 16..23}; lane>=16: K in {8..15, 24..31}
    float zf[32];
    v16bf a0, a1;
    #pragma unroll
    for (int e = 0; e < 16; ++e) {
      const int c = ((e >> 3) << 4) + (hf << 3) + (e & 7);
      zf[e]      = __builtin_nontemporal_load(zb + (size_t)c * HW);
      zf[e + 16] = __builtin_nontemporal_load(zb + (size_t)(c + 32) * HW);
      a0[e] = (__bf16)zf[e];
      a1[e] = (__bf16)zf[e + 16];
    }

    // ---- prefetch next m-tile's A data (one full codebook sweep of lead) ----
    if (j + 1 < MT_PER_WAVE) {
      const float* zn = zb + 16;    // next tile: same batch, p += 16
      #pragma unroll
      for (int e = 0; e < 16; ++e) {
        const int c = ((e >> 3) << 4) + (hf << 3) + (e & 7);
        __builtin_prefetch(zn + (size_t)c * HW, 0, 1);
        __builtin_prefetch(zn + (size_t)(c + 32) * HW, 0, 1);
      }
    }

    float bestV[8]; int bestI[8];
    #pragma unroll
    for (int i = 0; i < 8; ++i) { bestV[i] = 3.4e38f; bestI[i] = 0; }

    auto loadB = [&](int t, ABu& b0, ABu& b1, float& en) {
      const int n = t * 16 + l16;                 // this lane's code column
      const char* rowp = (const char*)lds_emb + n * (ROW_STRIDE * 2);
      b0.h[0] = *(const bf16x8*)(rowp + hf * 16);
      b0.h[1] = *(const bf16x8*)(rowp + 32 + hf * 16);
      b1.h[0] = *(const bf16x8*)(rowp + 64 + hf * 16);
      b1.h[1] = *(const bf16x8*)(rowp + 96 + hf * 16);
      en = lds_norm[n];
    };
    // n = 16t + l16 is strictly increasing per lane, so strict '<' already
    // keeps the first (lowest-index) minimum: no tie-break needed here.
    auto score = [&](const v8f& acc, float en, int t) {
      const int n = t * 16 + l16;
      #pragma unroll
      for (int i = 0; i < 8; ++i) {               // C slot i -> row M = i + 8*hf
        const float s = fmaf(-2.0f, acc[i], en);
        if (s < bestV[i]) { bestV[i] = s; bestI[i] = n; }
      }
    };

    // ---- codebook sweep, double-buffered B fragments (2 WMMAs per tile) ----
    ABu b0A, b1A, b0B, b1B;
    float enA, enB;
    loadB(0, b0A, b1A, enA);
    for (int t = 0; t < NT; t += 2) {
      loadB(t + 1, b0B, b1B, enB);                // prefetch odd tile
      v8f accA = {};
      accA = __builtin_amdgcn_wmma_f32_16x16x32_bf16(false, a0, false, b0A.v,
                                                     (short)0, accA, false, false);
      accA = __builtin_amdgcn_wmma_f32_16x16x32_bf16(false, a1, false, b1A.v,
                                                     (short)0, accA, false, false);
      const float enA0 = enA;
      if (t + 2 < NT) loadB(t + 2, b0A, b1A, enA);  // prefetch next even tile
      score(accA, enA0, t);
      v8f accB = {};
      accB = __builtin_amdgcn_wmma_f32_16x16x32_bf16(false, a0, false, b0B.v,
                                                     (short)0, accB, false, false);
      accB = __builtin_amdgcn_wmma_f32_16x16x32_bf16(false, a1, false, b1B.v,
                                                     (short)0, accB, false, false);
      score(accB, enB, t + 1);
    }

    // ---- per-row argmin: reduce over the 16 lanes of each half (tie-break
    //      on index here, where different n compete) ----
    #pragma unroll
    for (int m = 8; m >= 1; m >>= 1) {
      #pragma unroll
      for (int i = 0; i < 8; ++i) {
        const float ov = __shfl_xor(bestV[i], m, 16);
        const int   oi = __shfl_xor(bestI[i], m, 16);
        if (ov < bestV[i] || (ov == bestV[i] && oi < bestI[i])) {
          bestV[i] = ov; bestI[i] = oi;
        }
      }
    }
    if (l16 == 0) {
      #pragma unroll
      for (int i = 0; i < 8; ++i) {
        const int r = i + 8 * hf;
        lds_idx[wave * 16 + r] = bestI[i];
        idx_out[rowbase + r]   = bestI[i];
      }
    }
    // LDS ops from the same wave are in-order; safe same-wave RAW.
    const int myIdx = lds_idx[wave * 16 + l16];   // code for this lane's row

    // ---- gather z_q (fp32 from L2-resident emb), store, accumulate loss ----
    const float* er = emb + myIdx * EMB_DIM;
    float* zo = z_q_out + zoff;
    #pragma unroll
    for (int e = 0; e < 16; ++e) {
      const int c = ((e >> 3) << 4) + (hf << 3) + (e & 7);
      const float q0 = er[c];
      const float q1 = er[c + 32];
      __builtin_nontemporal_store(q0, zo + (size_t)c * HW);        // z_q_st == z_q
      __builtin_nontemporal_store(q1, zo + (size_t)(c + 32) * HW);
      const float d0 = q0 - zf[e];
      const float d1 = q1 - zf[e + 16];
      lsum += d0 * d0 + d1 * d1;
    }
  }

  // ---- deterministic block reduction of squared-error sum ----
  #pragma unroll
  for (int m = 16; m >= 1; m >>= 1) lsum += __shfl_xor(lsum, m, 32);
  if (lane == 0) lds_red[wave] = lsum;
  __syncthreads();
  if (tid == 0) {
    float s = 0.f;
    #pragma unroll
    for (int w = 0; w < WAVES_PER_BLOCK; ++w) s += lds_red[w];
    partials[blockIdx.x] = s;
  }
}

// Deterministic final reduction of the 256 block partials -> scalar loss.
__global__ __launch_bounds__(256, 1)
void vq_loss_reduce_kernel(const float* __restrict__ partials,
                           float* __restrict__ loss_out)
{
  __shared__ float sb[256];
  const int tid = threadIdx.x;
  sb[tid] = partials[tid];
  __syncthreads();
  for (int m = 128; m >= 1; m >>= 1) {
    if (tid < m) sb[tid] += sb[tid + m];
    __syncthreads();
  }
  if (tid == 0) loss_out[0] = sb[0] * (LOSS_W / (float)NUMEL);
}

extern "C" void kernel_launch(void* const* d_in, const int* in_sizes, int n_in,
                              void* d_out, int out_size, void* d_ws, size_t ws_size,
                              hipStream_t stream)
{
  const float* z_e = (const float*)d_in[0];   // (32, 64, 64, 64) fp32
  const float* emb = (const float*)d_in[1];   // (1024, 64) fp32

  float* out      = (float*)d_out;
  float* z_q_out  = out;                      // 8388608 floats (b,c,h,w)
  float* loss_out = out + NUMEL;              // 1 float
  int*   idx_out  = (int*)(out + NUMEL + 1);  // 131072 indices
  float* partials = (float*)d_ws;             // 256 floats of scratch

  const size_t smem = LDS_EMB_BYTES + LDS_NORM_BYTES +
                      WAVES_PER_BLOCK * 16 * 4 + WAVES_PER_BLOCK * 4;

  vq_fused_kernel<<<GRID1, 256, smem, stream>>>(z_e, emb, z_q_out, idx_out,
                                                partials);
  vq_loss_reduce_kernel<<<1, 256, 0, stream>>>(partials, loss_out);
}
